// DeformConv_85083302134219
// MI455X (gfx1250) — compile-verified, matrix operand order checked by reading, not run on hardware
//
#include <hip/hip_runtime.h>
#include <hip/hip_bf16.h>

// ---------------------------------------------------------------------------
// Fused DCNv2 + BN + ReLU for MI455X (gfx1250, wave32, WMMA).
// B=1, C=128, O=128, H=152, W=272, K=9.
//
// Block = 256 threads (8 waves), one 128-pixel row segment:
//   p1: offset conv (27ch)                -> LDS
//   p2: bilinear corner metadata          -> LDS
//   p3: 4 chunks x 32 channels: stage A (bf16 weight) + B (sampled mod, bf16)
//       in LDS; GEMM with v_wmma_f32_16x16x32_bf16, f32 accumulate.
//       Each wave owns 2 M-tiles x 4 N-tiles -> 12 ds_load_b128 per 8 WMMAs.
//   p4: folded BN + ReLU, coalesced f32 stores.
// mod (190MB logical intermediate) never touches HBM.
// ---------------------------------------------------------------------------

typedef __attribute__((ext_vector_type(16))) __bf16    v16bf;
typedef __attribute__((ext_vector_type(8)))  float     v8f;
typedef __attribute__((ext_vector_type(4)))  unsigned  v4u;

#define C_IN   128
#define O_OUT  128
#define H_DIM  152
#define W_DIM  272
#define HW     (H_DIM * W_DIM)
#define KTAPS  9
#define KTOT   (C_IN * KTAPS)   // 1152
#define NT     128              // pixel tile (8 N-tiles of 16)
#define CHC    32               // channels per chunk
#define KC     (CHC * KTAPS)    // 288 = 9 k-steps of 32
#define KCP    296              // LDS K stride: 592B rows -> 16B aligned, 148-dword
                                // stride: lane banks i*20 mod 64, conflict-free b128

__device__ __forceinline__ unsigned short f2bf(float f) {
  unsigned u = __builtin_bit_cast(unsigned, f);
  u += 0x7FFFu + ((u >> 16) & 1u);               // RNE fp32->bf16
  return (unsigned short)(u >> 16);
}

// 16-bit A/B fragment (ISA 7.12.2): lane<16 holds K{0..7,16..23} of row lane&15,
// lane>=16 holds K{8..15,24..31}. Each half = 4 consecutive dwords -> 2x ds_load_b128.
__device__ __forceinline__ v16bf load_frag(const unsigned short* rowbase, int kb, int kh) {
  union { v16bf v; v4u q[2]; } f;
  const v4u* p = (const v4u*)(rowbase + kb + (kh << 3));  // 16B aligned (see KCP)
  f.q[0] = p[0];   // K = base+0..7   (pairs {0,2,4,6})
  f.q[1] = p[2];   // K = base+16..23 (pairs {16,18,20,22})
  return f.v;
}

// ---- pre-kernel: weight f32 -> bf16 into workspace (147456 elems) ----------
__global__ __launch_bounds__(256) void wconv_bf16(const float* __restrict__ wgt,
                                                  unsigned short* __restrict__ wbf) {
  int i = (blockIdx.x * 256 + threadIdx.x) * 4;   // 147456 % 4 == 0
  if (i < O_OUT * KTOT) {
    float4 v = *(const float4*)(wgt + i);
    unsigned long long pk =
        (unsigned long long)f2bf(v.x)
      | ((unsigned long long)f2bf(v.y) << 16)
      | ((unsigned long long)f2bf(v.z) << 32)
      | ((unsigned long long)f2bf(v.w) << 48);
    *(unsigned long long*)(wbf + i) = pk;
  }
}

__global__ __launch_bounds__(256) void dcn_bn_relu_fused(
    const float* __restrict__ x,     const float* __restrict__ offw,
    const float* __restrict__ offb,  const float* __restrict__ wgt,
    const float* __restrict__ bias,  const float* __restrict__ gamma,
    const float* __restrict__ beta,  const float* __restrict__ rmean,
    const float* __restrict__ rvar,  const unsigned short* __restrict__ wbf,
    int usews, float* __restrict__ out)
{
  extern __shared__ char smem[];
  float* om2  = (float*)smem;                         // [2][128][27]
  float* cw   = om2 + 2 * 128 * 27;                   // [4][1152] corner weights
  int*   cc   = (int*)(cw + 4 * 1152);                // [4][1152] clamped coords
  float* bnsc = (float*)(cc + 4 * 1152);              // [128]
  float* bnsh = bnsc + 128;                           // [128]
  unsigned short* Al = (unsigned short*)(bnsh + 128); // [128][KCP] bf16 weight tile
  unsigned short* Bl = Al + 128 * KCP;                // [128][KCP] bf16 mod tile

  const int tid = threadIdx.x;
  const int h   = blockIdx.y;
  const int w0  = blockIdx.x * NT;

  if (tid < 128) {
    float sc  = gamma[tid] * rsqrtf(rvar[tid] + 1e-5f);
    bnsc[tid] = sc;
    bnsh[tid] = (bias[tid] - rmean[tid]) * sc + beta[tid];
  }

  // ---- phase 1: offset conv (27 ch); two thread-halves split input channels
  {
    const int px    = tid & 127;
    const int halfc = tid >> 7;
    const int w     = w0 + px;
    float acc[27];
#pragma unroll
    for (int i = 0; i < 27; ++i) acc[i] = 0.f;
    if (w < W_DIM) {
#pragma unroll 1
      for (int c = halfc * 64; c < halfc * 64 + 64; ++c) {
        const float* xc = x + (size_t)c * HW;
#pragma unroll
        for (int ky = 0; ky < 3; ++ky) {
          int yy = h + ky - 1;
          if ((unsigned)yy >= (unsigned)H_DIM) continue;
#pragma unroll
          for (int kx = 0; kx < 3; ++kx) {
            int xx = w + kx - 1;
            if ((unsigned)xx >= (unsigned)W_DIM) continue;
            float xv = xc[yy * W_DIM + xx];
            const float* wp = offw + c * 9 + ky * 3 + kx;  // uniform -> s_load
#pragma unroll
            for (int oc = 0; oc < 27; ++oc)
              acc[oc] = fmaf(xv, wp[oc * KTOT], acc[oc]);
          }
        }
      }
    }
#pragma unroll
    for (int oc = 0; oc < 27; ++oc) om2[(halfc * 128 + px) * 27 + oc] = acc[oc];
  }
  __syncthreads();
  for (int i = tid; i < 128 * 27; i += 256)
    om2[i] = om2[i] + om2[128 * 27 + i] + offb[i % 27];
  __syncthreads();

  // ---- phase 2: corner metadata; dy_k = om[2k], dx_k = om[2k+1], mask = sigma(om[18+k])
  for (int i = tid; i < 1152; i += 256) {
    int px = i & 127, k = i >> 7;
    const float* o = om2 + px * 27;
    float dy = o[2 * k], dx = o[2 * k + 1];
    float mk = 1.f / (1.f + __expf(-o[18 + k]));
    int w = w0 + px;
    float py  = (float)(h + k / 3 - 1) + dy;
    float pxx = (float)(w + (k % 3) - 1) + dx;
    float y0f = floorf(py), x0f = floorf(pxx);
    float ly = py - y0f, lx = pxx - x0f;
    int y0 = (int)y0f, x0 = (int)x0f;
    int y1 = y0 + 1,   x1 = x0 + 1;
    float vy0 = (y0 >= 0 && y0 < H_DIM) ? 1.f : 0.f;
    float vy1 = (y1 >= 0 && y1 < H_DIM) ? 1.f : 0.f;
    float vx0 = (x0 >= 0 && x0 < W_DIM) ? 1.f : 0.f;
    float vx1 = (x1 >= 0 && x1 < W_DIM) ? 1.f : 0.f;
    float m   = (w < W_DIM) ? mk : 0.f;
    cw[i]            = (1.f - ly) * (1.f - lx) * vy0 * vx0 * m;
    cw[1152 + i]     = (1.f - ly) * lx         * vy0 * vx1 * m;
    cw[2 * 1152 + i] = ly * (1.f - lx)         * vy1 * vx0 * m;
    cw[3 * 1152 + i] = ly * lx                 * vy1 * vx1 * m;
    cc[i]            = min(max(y0, 0), H_DIM - 1);
    cc[1152 + i]     = min(max(y1, 0), H_DIM - 1);
    cc[2 * 1152 + i] = min(max(x0, 0), W_DIM - 1);
    cc[3 * 1152 + i] = min(max(x1, 0), W_DIM - 1);
  }

  // ---- phase 3: WMMA GEMM. Wave w: M-pair mp = w&3, N-quad q = w>>2.
  const int lane = tid & 31, wave = tid >> 5;
  const int l15 = lane & 15, kh = lane >> 4;
  const int mp = wave & 3, q = wave >> 2;
  v8f acc[2][4];
#pragma unroll
  for (int mi = 0; mi < 2; ++mi)
#pragma unroll
    for (int n = 0; n < 4; ++n) acc[mi][n] = (v8f){0, 0, 0, 0, 0, 0, 0, 0};

  const unsigned short* arow0 = Al + (mp * 32 + l15) * KCP;
  const unsigned short* arow1 = arow0 + 16 * KCP;

#pragma unroll 1
  for (int chunk = 0; chunk < 4; ++chunk) {
    __syncthreads();   // previous chunk's WMMA reads done before overwrite
    // A: bf16 weight chunk -> LDS. 16B global loads + 16B LDS stores when
    // pre-converted weights are in workspace; else f32 + convert fallback.
    if (usews) {
      const unsigned short* ws0 = wbf + chunk * KC;
      for (int i = tid; i < 128 * (KC / 8); i += 256) {          // 4608/256 = 18
        int o = i / (KC / 8), k8 = i - o * (KC / 8);
        *(v4u*)(Al + o * KCP + k8 * 8) =
            *(const v4u*)(ws0 + (size_t)o * KTOT + k8 * 8);
      }
    } else {
      const float* ws0 = wgt + chunk * KC;
      for (int i = tid; i < 128 * (KC / 4); i += 256) {          // 9216/256 = 36
        int o = i / (KC / 4), k4 = i - o * (KC / 4);
        float4 v = *(const float4*)(ws0 + (size_t)o * KTOT + k4 * 4);
        unsigned long long pk =
            (unsigned long long)f2bf(v.x)
          | ((unsigned long long)f2bf(v.y) << 16)
          | ((unsigned long long)f2bf(v.z) << 32)
          | ((unsigned long long)f2bf(v.w) << 48);
        *(unsigned long long*)(Al + o * KCP + k4 * 4) = pk;      // 8B aligned
      }
    }
    // B: bilinear sample + modulate 32 channels (gathers are L2-resident: x=21MB)
    {
      const float* xb = x + (size_t)(chunk * CHC) * HW;
      for (int i = tid; i < 1152; i += 256) {
        int px = i & 127, k = i >> 7;
        float w00 = cw[i], w01 = cw[1152 + i], w10 = cw[2 * 1152 + i], w11 = cw[3 * 1152 + i];
        int y0W = cc[i] * W_DIM, y1W = cc[1152 + i] * W_DIM;
        int x0c = cc[2 * 1152 + i], x1c = cc[3 * 1152 + i];
        int o00 = y0W + x0c, o01 = y0W + x1c, o10 = y1W + x0c, o11 = y1W + x1c;
        unsigned short* brow = Bl + px * KCP + k;
#pragma unroll 4
        for (int c = 0; c < CHC; ++c) {
          const float* xc = xb + (size_t)c * HW;
          float v = w00 * xc[o00] + w01 * xc[o01] + w10 * xc[o10] + w11 * xc[o11];
          brow[c * 9] = f2bf(v);     // K index = c_local*9 + tap
        }
      }
    }
    __syncthreads();
    // 9 k-steps of 32: 4 b128 (2 A frags) + 8 b128 (4 B frags) per 8 WMMAs
#pragma unroll 1
    for (int ks = 0; ks < KC / 32; ++ks) {
      int kb = ks * 32;
      v16bf a0 = load_frag(arow0, kb, kh);
      v16bf a1 = load_frag(arow1, kb, kh);
#pragma unroll
      for (int n = 0; n < 4; ++n) {
        v16bf b = load_frag(Bl + (q * 64 + n * 16 + l15) * KCP, kb, kh);
        acc[0][n] = __builtin_amdgcn_wmma_f32_16x16x32_bf16(
            false, a0, false, b, (short)0, acc[0][n], false, false);
        acc[1][n] = __builtin_amdgcn_wmma_f32_16x16x32_bf16(
            false, a1, false, b, (short)0, acc[1][n], false, false);
      }
    }
  }

  // ---- phase 4: BN + ReLU. C/D layout: VGPR r -> M = r + 8*kh, lane&15 -> N.
#pragma unroll
  for (int mi = 0; mi < 2; ++mi) {
#pragma unroll
    for (int n = 0; n < 4; ++n) {
      int w = w0 + q * 64 + n * 16 + l15;
      if (w < W_DIM) {
#pragma unroll
        for (int r = 0; r < 8; ++r) {
          int o = mp * 32 + mi * 16 + kh * 8 + r;
          float v = acc[mi][n][r] * bnsc[o] + bnsh[o];
          out[(size_t)o * HW + (size_t)h * W_DIM + w] = v > 0.f ? v : 0.f;
        }
      }
    }
  }
}

extern "C" void kernel_launch(void* const* d_in, const int* in_sizes, int n_in,
                              void* d_out, int out_size, void* d_ws, size_t ws_size,
                              hipStream_t stream) {
  (void)in_sizes; (void)n_in; (void)out_size;
  const float* x     = (const float*)d_in[0];
  const float* offw  = (const float*)d_in[1];
  const float* offb  = (const float*)d_in[2];
  const float* wgt   = (const float*)d_in[3];
  const float* bias  = (const float*)d_in[4];
  const float* gam   = (const float*)d_in[5];
  const float* bet   = (const float*)d_in[6];
  const float* rmean = (const float*)d_in[7];
  const float* rvar  = (const float*)d_in[8];
  float* out = (float*)d_out;

  unsigned short* wbf = (unsigned short*)d_ws;
  const size_t need = (size_t)O_OUT * KTOT * sizeof(unsigned short);  // 294912 B
  int usews = (d_ws != nullptr && ws_size >= need) ? 1 : 0;
  if (usews) {
    int n4 = (O_OUT * KTOT) / 4;                       // 36864
    wconv_bf16<<<(n4 + 255) / 256, 256, 0, stream>>>(wgt, wbf);
  }

  // LDS: 65536 B metadata + 2*128*296*2 = 151552 B bf16 tiles = 217088 B
  // (< 320 KB per WGP; one 8-wave workgroup per WGP)
  size_t smem = (size_t)(2 * 128 * 27 + 4 * 1152 + 4 * 1152 + 256) * 4
              + (size_t)2 * 128 * KCP * 2;
  dim3 grid((W_DIM + NT - 1) / NT, H_DIM);             // 3 x 152
  hipLaunchKernelGGL(dcn_bn_relu_fused, grid, dim3(256), smem, stream,
                     x, offw, offb, wgt, bias, gam, bet, rmean, rvar, wbf, usews, out);
}
